// GNNClassifier_74182675136829
// MI455X (gfx1250) — matrix operand order repaired and checked by reading, compiled, not run on hardware
//
#include <hip/hip_runtime.h>

// ---------------------------------------------------------------------------
// Types for gfx1250 WMMA (wave32): v_wmma_f32_16x16x32_bf16
// ---------------------------------------------------------------------------
typedef __bf16 bf16_t;
typedef __attribute__((ext_vector_type(8)))  bf16_t bf16x8;
typedef __attribute__((ext_vector_type(16))) bf16_t bf16x16;
typedef __attribute__((ext_vector_type(8)))  float  f32x8;

static __device__ __forceinline__ unsigned bfround(unsigned u) {
  return u + 0x7FFFu + ((u >> 16) & 1u);  // RNE f32->bf16 in the high half
}
static __device__ __forceinline__ bf16_t f2bf(float f) {
  return __builtin_bit_cast(bf16_t,
      (unsigned short)(bfround(__float_as_uint(f)) >> 16));
}

// order-preserving float <-> uint for atomicMax-based segment max
static __device__ __forceinline__ unsigned f2ord(float f) {
  unsigned u = __float_as_uint(f);
  return (u & 0x80000000u) ? ~u : (u | 0x80000000u);
}
static __device__ __forceinline__ float ord2f(unsigned k) {
  unsigned u = (k & 0x80000000u) ? (k ^ 0x80000000u) : ~k;
  return __uint_as_float(u);
}

// ---------------------------------------------------------------------------
// bf16 GEMM with async-to-LDS staging (CDNA5 ASYNCcnt path).
// All operands are pre-padded bf16: A has >= gridDim.x*64 rows, B has
// >= Kpad rows and >= gridDim.y*64 cols, Kpad % 64 == 0.  The staging
// loop has NO guards and NO VGPR round-trip: global_load_async_to_lds_b128.
// Block: 256 threads = 8 waves, tile 64(M) x 64(N), K stepped by 64.
// ---------------------------------------------------------------------------
__global__ __launch_bounds__(256)
void gemm_bf16_wmma(const bf16_t* __restrict__ A, int lda,
                    const bf16_t* __restrict__ B, int ldb,
                    const float* __restrict__ bias,
                    float* __restrict__ C, int ldc,
                    int M, int N, int Kpad)
{
  __shared__ __align__(16) bf16_t As[64][72];  // [m][k], 144B row stride
  __shared__ __align__(16) bf16_t Bs[64][72];  // [k][n]

  const int t    = threadIdx.x;
  const int lane = t & 31;
  const int wave = t >> 5;
  const int wm   = wave & 3;
  const int wn   = wave >> 2;
  const int m0   = blockIdx.x * 64;
  const int n0   = blockIdx.y * 64;
  const int row  = lane & 15;   // A row / C column within 16
  const int hi   = lane >> 4;   // lane half select

  f32x8 acc0 = {0.f, 0.f, 0.f, 0.f, 0.f, 0.f, 0.f, 0.f};
  f32x8 acc1 = acc0;

  // per-thread copy slots: 512 x 16B chunks per 64x64 bf16 slab, 2 each
  const int r0 = t >> 3;              // chunk row for slot 0 (0..31)
  const int c0 = (t & 7) * 8;         // chunk col in elements

  for (int kk = 0; kk < Kpad; kk += 64) {
#pragma unroll
    for (int i = 0; i < 2; ++i) {
      int ar = r0 + i * 32;
      // A slab row ar, cols [kk+c0, kk+c0+8)
      {
        const bf16_t* ga = &A[(size_t)(m0 + ar) * lda + kk + c0];
        unsigned      la = (unsigned)(size_t)&As[ar][c0];
        asm volatile("global_load_async_to_lds_b128 %0, %1, off"
                     :: "v"(la), "v"(ga) : "memory");
      }
      // B slab row kk+ar, cols [n0+c0, n0+c0+8)
      {
        const bf16_t* gb = &B[(size_t)(kk + ar) * ldb + n0 + c0];
        unsigned      lb = (unsigned)(size_t)&Bs[ar][c0];
        asm volatile("global_load_async_to_lds_b128 %0, %1, off"
                     :: "v"(lb), "v"(gb) : "memory");
      }
    }
    asm volatile("s_wait_asynccnt 0" ::: "memory");
    __syncthreads();

#pragma unroll
    for (int ks = 0; ks < 64; ks += 32) {
      // A fragment (ISA 16-bit A layout): lanes0-15 K{0..7,16..23},
      // lanes16-31 K{8..15,24..31}; two contiguous ds_load_b128.
      bf16x8 a0 = *(const bf16x8*)&As[wm * 16 + row][ks + hi * 8];
      bf16x8 a1 = *(const bf16x8*)&As[wm * 16 + row][ks + 16 + hi * 8];
      bf16x16 af = __builtin_shufflevector(a0, a1,
          0, 1, 2, 3, 4, 5, 6, 7, 8, 9, 10, 11, 12, 13, 14, 15);

      // B fragments: lane holds column N=lane%16, K in same order as A.
      bf16x16 bf0, bf1;
#pragma unroll
      for (int s = 0; s < 16; ++s) {
        int kb = ks + hi * 8 + (s & 7) + (s >> 3) * 16;
        bf0[s] = Bs[kb][wn * 32 + row];
        bf1[s] = Bs[kb][wn * 32 + 16 + row];
      }

      acc0 = __builtin_amdgcn_wmma_f32_16x16x32_bf16(
          false, af, false, bf0, (short)0, acc0, false, false);
      acc1 = __builtin_amdgcn_wmma_f32_16x16x32_bf16(
          false, af, false, bf1, (short)0, acc1, false, false);
    }
    __syncthreads();
  }

  // C/D layout: lane = N (mod 16), VGPR v -> M = hi*8 + v
#pragma unroll
  for (int v = 0; v < 8; ++v) {
    int gm = m0 + wm * 16 + hi * 8 + v;
    if (gm < M) {
      int gn0 = n0 + wn * 32 + row;
      int gn1 = gn0 + 16;
      if (gn0 < N) {
        float r = acc0[v];
        if (bias) r += bias[gn0];
        C[(size_t)gm * ldc + gn0] = r;
      }
      if (gn1 < N) {
        float r = acc1[v];
        if (bias) r += bias[gn1];
        C[(size_t)gm * ldc + gn1] = r;
      }
    }
  }
}

// ---------------------------------------------------------------------------
// fp32 -> bf16 with row padding: dst[padRows, cols], rows >= srcRows zeroed
// ---------------------------------------------------------------------------
__global__ void cvt_bf16_pad(const float* __restrict__ src, bf16_t* __restrict__ dst,
                             int rows, int cols, int padRows)
{
  size_t i = (size_t)blockIdx.x * blockDim.x + threadIdx.x;
  if (i >= (size_t)padRows * cols) return;
  int n = (int)(i / cols);
  float v = (n < rows) ? src[i] : 0.f;
  dst[i] = f2bf(v);
}

// ---------------------------------------------------------------------------
// Weight fuse: Wcat[K,512](bf16) = [W0 | W1 | W2 | Ws | 0pad], bcat=[0x252,bs]
// ---------------------------------------------------------------------------
__global__ void pack_weights(const float* __restrict__ W0, const float* __restrict__ W1,
                             const float* __restrict__ W2, const float* __restrict__ Wsf,
                             const float* __restrict__ bs,
                             bf16_t* __restrict__ Wcat, float* __restrict__ bcat, int K)
{
  int i = blockIdx.x * blockDim.x + threadIdx.x;
  if (i < 508) bcat[i] = (i < 252) ? 0.f : bs[i - 252];
  if (i >= K * 512) return;
  int k = i >> 9, c = i & 511;
  float v = 0.f;
  if      (c < 84)  v = W0[k * 84 + c];
  else if (c < 168) v = W1[k * 84 + (c - 84)];
  else if (c < 252) v = W2[k * 84 + (c - 168)];
  else if (c < 508) v = Wsf[k * 256 + (c - 252)];
  Wcat[i] = f2bf(v);
}

// ---------------------------------------------------------------------------
// Attention logits: al_s[n,h] = sum_d h[n,h,d]*a_s[h,d]  (and a_d)
// h is a strided view (row stride hld) into hfeat
// ---------------------------------------------------------------------------
__global__ void attn_logits(const float* __restrict__ h, int hld,
                            const float* __restrict__ a_s,
                            const float* __restrict__ a_d,
                            float* __restrict__ al_s,
                            float* __restrict__ al_d, int Nn)
{
  int i = blockIdx.x * blockDim.x + threadIdx.x;
  if (i >= Nn * 2) return;
  int n = i >> 1, hh = i & 1;
  const float* hp = h + (size_t)n * hld + hh * 42;
  const float* as = a_s + hh * 42;
  const float* ad = a_d + hh * 42;
  float ss = 0.f, sd = 0.f;
#pragma unroll 6
  for (int d = 0; d < 42; ++d) {
    float v = hp[d];
    ss += v * as[d];
    sd += v * ad[d];
  }
  al_s[i] = ss;
  al_d[i] = sd;
}

// ---------------------------------------------------------------------------
// Per-order scratch init: segment-max keys to ord(-inf), den=0, out=0
// ---------------------------------------------------------------------------
__global__ void init_order(unsigned* __restrict__ m, float* __restrict__ den,
                           float* __restrict__ outo, int Nn)
{
  size_t i = (size_t)blockIdx.x * blockDim.x + threadIdx.x;
  if (i < (size_t)Nn * 2) { m[i] = 0x007FFFFFu; den[i] = 0.f; }  // ~bits(-inf)
  if (i < (size_t)Nn * 84) outo[i] = 0.f;
}

static __device__ __forceinline__ void edge_sd(int e, int E,
                                               const int* src, const int* dst,
                                               int& s, int& d)
{
  if (e < E) { s = src[e]; d = dst[e]; }
  else       { s = e - E;  d = e - E; }  // self loop
}

__global__ void edge_max(const int* __restrict__ src, const int* __restrict__ dst,
                         int E, int Nn,
                         const float* __restrict__ al_s,
                         const float* __restrict__ al_d,
                         unsigned* __restrict__ m)
{
  int e = blockIdx.x * blockDim.x + threadIdx.x;
  if (e >= E + Nn) return;
  int s, d;
  edge_sd(e, E, src, dst, s, d);
#pragma unroll
  for (int hh = 0; hh < 2; ++hh) {
    float v = al_s[(size_t)s * 2 + hh] + al_d[(size_t)d * 2 + hh];
    v = v > 0.f ? v : 0.2f * v;  // leaky_relu(0.2)
    atomicMax(&m[(size_t)d * 2 + hh], f2ord(v));
  }
}

__global__ void edge_expsum(const int* __restrict__ src, const int* __restrict__ dst,
                            int E, int Nn,
                            const float* __restrict__ al_s,
                            const float* __restrict__ al_d,
                            const unsigned* __restrict__ m,
                            float* __restrict__ exb, float* __restrict__ den)
{
  int e = blockIdx.x * blockDim.x + threadIdx.x;
  if (e >= E + Nn) return;
  int s, d;
  edge_sd(e, E, src, dst, s, d);
#pragma unroll
  for (int hh = 0; hh < 2; ++hh) {
    float v = al_s[(size_t)s * 2 + hh] + al_d[(size_t)d * 2 + hh];
    v = v > 0.f ? v : 0.2f * v;
    float mx = ord2f(m[(size_t)d * 2 + hh]);
    float x = __expf(v - mx);
    exb[(size_t)e * 2 + hh] = x;
    atomicAdd(&den[(size_t)d * 2 + hh], x);
  }
}

// one thread per (edge, channel); blockIdx.y = channel 0..83
__global__ void edge_aggr(const int* __restrict__ src, const int* __restrict__ dst,
                          int E, int Nn,
                          const float* __restrict__ exb,
                          const float* __restrict__ den,
                          const float* __restrict__ h, int hld,
                          float* __restrict__ outo)
{
  int e = blockIdx.x * blockDim.x + threadIdx.x;
  if (e >= E + Nn) return;
  int c = blockIdx.y;
  int s, d;
  edge_sd(e, E, src, dst, s, d);
  int hh = (c >= 42) ? 1 : 0;
  float alpha = exb[(size_t)e * 2 + hh] / den[(size_t)d * 2 + hh];
  atomicAdd(&outo[(size_t)d * 84 + c], alpha * h[(size_t)s * hld + c]);
}

// ---------------------------------------------------------------------------
// hcatb[n, 0:512](bf16) = [out0+b0 | out1+b1 | out2+b2 | hfeat[:,252:508] | 0]
// rows >= Nn zero-padded (self columns already have bs from fused GEMM)
// ---------------------------------------------------------------------------
__global__ void concat_bias_bf16(const float* __restrict__ o0, const float* __restrict__ o1,
                                 const float* __restrict__ o2,
                                 const float* __restrict__ b0, const float* __restrict__ b1,
                                 const float* __restrict__ b2,
                                 const float* __restrict__ hfeat,
                                 bf16_t* __restrict__ hcatb, int Nn, int padRows)
{
  size_t i = (size_t)blockIdx.x * blockDim.x + threadIdx.x;
  if (i >= (size_t)padRows * 512) return;
  int n = (int)(i >> 9), c = (int)(i & 511);
  float v = 0.f;
  if (n < Nn) {
    if      (c < 84)  v = o0[(size_t)n * 84 + c]         + b0[c];
    else if (c < 168) v = o1[(size_t)n * 84 + (c - 84)]  + b1[c - 84];
    else if (c < 252) v = o2[(size_t)n * 84 + (c - 168)] + b2[c - 168];
    else if (c < 508) v = hfeat[(size_t)n * 508 + c];
  }
  hcatb[i] = f2bf(v);
}

// in-place layernorm (ddof=0) + affine + ReLU, one block (256 thr) per node
__global__ __launch_bounds__(256)
void layernorm_relu(float* __restrict__ h, const float* __restrict__ g,
                    const float* __restrict__ be)
{
  __shared__ float red[256];
  int n = blockIdx.x, c = threadIdx.x;
  float v = h[(size_t)n * 256 + c];
  red[c] = v; __syncthreads();
  for (int s = 128; s > 0; s >>= 1) { if (c < s) red[c] += red[c + s]; __syncthreads(); }
  float mu = red[0] * (1.f / 256.f);
  __syncthreads();
  float dv = v - mu;
  red[c] = dv * dv; __syncthreads();
  for (int s = 128; s > 0; s >>= 1) { if (c < s) red[c] += red[c + s]; __syncthreads(); }
  float var = red[0] * (1.f / 256.f);
  float y = dv * rsqrtf(var + 1e-5f) * g[c] + be[c];
  h[(size_t)n * 256 + c] = y > 0.f ? y : 0.f;
}

// ---------------------------------------------------------------------------
// Heads: tl = h@Wt+bt (6), il = h@Wi+bi (8)  -> d_out[0 : N*6), [N*6 : N*14)
// ---------------------------------------------------------------------------
__global__ void heads_lin(const float* __restrict__ h,
                          const float* __restrict__ Wt, const float* __restrict__ bt,
                          const float* __restrict__ Wi, const float* __restrict__ bi,
                          float* __restrict__ out, int Nn)
{
  int i = blockIdx.x * blockDim.x + threadIdx.x;
  if (i >= Nn * 14) return;
  int n = i / 14, j = i % 14;
  const float* hp = h + (size_t)n * 256;
  if (j < 6) {
    float s = bt[j];
    for (int k = 0; k < 256; ++k) s += hp[k] * Wt[k * 6 + j];
    out[(size_t)n * 6 + j] = s;
  } else {
    int jj = j - 6;
    float s = bi[jj];
    for (int k = 0; k < 256; ++k) s += hp[k] * Wi[k * 8 + jj];
    out[(size_t)Nn * 6 + (size_t)n * 8 + jj] = s;
  }
}

// ml = [h | sigmoid(tl) | softmax(il)] @ Wm + bm  -> d_out[N*14 : N*24)
__global__ void heads_ml(const float* __restrict__ h,
                         const float* __restrict__ tlil,
                         const float* __restrict__ Wm, const float* __restrict__ bm,
                         float* __restrict__ ml, int Nn)
{
  int i = blockIdx.x * blockDim.x + threadIdx.x;
  if (i >= Nn * 10) return;
  int n = i / 10, j = i % 10;
  const float* tl = tlil + (size_t)n * 6;
  const float* il = tlil + (size_t)Nn * 6 + (size_t)n * 8;
  float mx = il[0];
#pragma unroll
  for (int k = 1; k < 8; ++k) mx = fmaxf(mx, il[k]);
  float ex[8]; float ssum = 0.f;
#pragma unroll
  for (int k = 0; k < 8; ++k) { ex[k] = __expf(il[k] - mx); ssum += ex[k]; }
  float s = bm[j];
  const float* hp = h + (size_t)n * 256;
  for (int k = 0; k < 256; ++k) s += hp[k] * Wm[k * 10 + j];
#pragma unroll
  for (int k = 0; k < 6; ++k)
    s += (1.f / (1.f + __expf(-tl[k]))) * Wm[(256 + k) * 10 + j];
#pragma unroll
  for (int k = 0; k < 8; ++k)
    s += (ex[k] / ssum) * Wm[(264 + k) * 10 + j];
  ml[(size_t)n * 10 + j] = s;
}

// ---------------------------------------------------------------------------
// Host orchestration
// d_in order: x, ei1, ei2, ei3, then params in dict insertion order:
//   per layer L in {0,1}: [W0,as0,ad0,b0, W1,as1,ad1,b1, W2,as2,ad2,b2,
//                          Ws,bs,Wp,bp,g,be]  (18 entries)
//   then Wt,bt,Wi,bi,Wm,bm  (indices 40..45)
// ---------------------------------------------------------------------------
extern "C" void kernel_launch(void* const* d_in, const int* in_sizes, int n_in,
                              void* d_out, int out_size, void* d_ws, size_t ws_size,
                              hipStream_t stream)
{
  (void)n_in; (void)out_size; (void)ws_size;

  const float* x = (const float*)d_in[0];
  const int* eib[3] = {(const int*)d_in[1], (const int*)d_in[2], (const int*)d_in[3]};
  int E[3] = { in_sizes[1] / 2, in_sizes[2] / 2, in_sizes[3] / 2 };
  const int Nn = in_sizes[0] / 768;
  const int Mpad = (Nn + 63) & ~63;

  // --- workspace carve-out ---
  char* ws = (char*)d_ws;
  size_t off = 0;
  auto take = [&](size_t bytes) -> void* {
    void* p = ws + off;
    off += (bytes + 255) & ~(size_t)255;
    return p;
  };
  float* out_o[3];  float* als[3];  float* ald[3];
  for (int o = 0; o < 3; ++o) out_o[o] = (float*)take((size_t)Nn * 84 * 4);
  for (int o = 0; o < 3; ++o) {
    als[o] = (float*)take((size_t)Nn * 2 * 4);
    ald[o] = (float*)take((size_t)Nn * 2 * 4);
  }
  unsigned* mbuf = (unsigned*)take((size_t)Nn * 2 * 4);
  float* den = (float*)take((size_t)Nn * 2 * 4);
  int Emax = E[0]; if (E[1] > Emax) Emax = E[1]; if (E[2] > Emax) Emax = E[2];
  float*  exb   = (float*)take(((size_t)Emax + Nn) * 2 * 4);
  float*  hfeat = (float*)take((size_t)Nn * 508 * 4);      // [h0|h1|h2|self] fp32
  bf16_t* xb    = (bf16_t*)take((size_t)Mpad * 768 * 2);   // bf16(x), row-padded
  bf16_t* hlayb = (bf16_t*)take((size_t)Mpad * 256 * 2);   // bf16(h1), row-padded
  bf16_t* hcatb = (bf16_t*)take((size_t)Mpad * 512 * 2);   // bf16 concat, 512-col pad
  bf16_t* wcatb = (bf16_t*)take((size_t)768 * 512 * 2);    // fused weights bf16
  bf16_t* wpb   = (bf16_t*)take((size_t)512 * 256 * 2);    // Wp bf16, 512-row pad
  float*  bcat  = (float*)take((size_t)508 * 4);           // fused bias
  float* hlay[2] = { (float*)take((size_t)Nn * 256 * 4),
                     (float*)take((size_t)Nn * 256 * 4) };

  const dim3 blk(256);

  // bf16 mirror of x (row-padded)
  cvt_bf16_pad<<<(unsigned)(((size_t)Mpad * 768 + 255) / 256), blk, 0, stream>>>(
      x, xb, Nn, 768, Mpad);

  int Kin = 768;
  for (int L = 0; L < 2; ++L) {
    int pb = 4 + L * 18;
    const float *Wg[3], *asg[3], *adg[3], *bg[3];
    for (int o = 0; o < 3; ++o) {
      Wg[o]  = (const float*)d_in[pb + 4 * o + 0];
      asg[o] = (const float*)d_in[pb + 4 * o + 1];
      adg[o] = (const float*)d_in[pb + 4 * o + 2];
      bg[o]  = (const float*)d_in[pb + 4 * o + 3];
    }
    const float* Ws = (const float*)d_in[pb + 12];
    const float* bs = (const float*)d_in[pb + 13];
    const float* Wp = (const float*)d_in[pb + 14];
    const float* bp = (const float*)d_in[pb + 15];
    const float* g  = (const float*)d_in[pb + 16];
    const float* be = (const float*)d_in[pb + 17];

    const bf16_t* Ab = (L == 0) ? xb : hlayb;

    // fuse [W0|W1|W2|Ws|0] -> Wcat (K x 512, bf16), bcat = [0 x252, bs]
    pack_weights<<<(unsigned)((Kin * 512 + 255) / 256), blk, 0, stream>>>(
        Wg[0], Wg[1], Wg[2], Ws, bs, wcatb, bcat, Kin);

    // one fused input GEMM: hfeat = A @ Wcat + bcat   (N = 508)
    {
      dim3 grid((unsigned)(Mpad / 64), 8u);
      gemm_bf16_wmma<<<grid, blk, 0, stream>>>(Ab, Kin, wcatb, 512, bcat,
                                               hfeat, 508, Nn, 508, Kin);
    }

    // attention logits over the strided h_o views
    for (int o = 0; o < 3; ++o)
      attn_logits<<<(unsigned)((Nn * 2 + 255) / 256), blk, 0, stream>>>(
          hfeat + o * 84, 508, asg[o], adg[o], als[o], ald[o], Nn);

    // edge-segmented softmax + aggregation (L2-resident atomics)
    for (int o = 0; o < 3; ++o) {
      int Et = E[o] + Nn;
      init_order<<<(unsigned)(((size_t)Nn * 84 + 255) / 256), blk, 0, stream>>>(
          mbuf, den, out_o[o], Nn);
      edge_max<<<(unsigned)((Et + 255) / 256), blk, 0, stream>>>(
          eib[o], eib[o] + E[o], E[o], Nn, als[o], ald[o], mbuf);
      edge_expsum<<<(unsigned)((Et + 255) / 256), blk, 0, stream>>>(
          eib[o], eib[o] + E[o], E[o], Nn, als[o], ald[o], mbuf, exb, den);
      dim3 gag((unsigned)((Et + 255) / 256), 84u);
      edge_aggr<<<gag, blk, 0, stream>>>(
          eib[o], eib[o] + E[o], E[o], Nn, exb, den, hfeat + o * 84, 508, out_o[o]);
    }

    // concat + order biases -> bf16 [Mpad,512]
    concat_bias_bf16<<<(unsigned)(((size_t)Mpad * 512 + 255) / 256), blk, 0, stream>>>(
        out_o[0], out_o[1], out_o[2], bg[0], bg[1], bg[2], hfeat, hcatb, Nn, Mpad);

    // Wp -> bf16 (512-row padded)
    cvt_bf16_pad<<<(unsigned)((512 * 256 + 255) / 256), blk, 0, stream>>>(
        Wp, wpb, 508, 256, 512);

    // projection  h = hcatb @ Wp + bp   (Kpad = 512, zero-padded)
    {
      dim3 grid((unsigned)(Mpad / 64), 4u);
      gemm_bf16_wmma<<<grid, blk, 0, stream>>>(hcatb, 512, wpb, 256, bp,
                                               hlay[L], 256, Nn, 256, 512);
    }
    // layernorm + affine + ReLU (in place)
    layernorm_relu<<<(unsigned)Nn, 256, 0, stream>>>(hlay[L], g, be);

    // bf16 mirror of layer output for next layer's GEMMs
    if (L == 0)
      cvt_bf16_pad<<<(unsigned)(((size_t)Mpad * 256 + 255) / 256), blk, 0, stream>>>(
          hlay[0], hlayb, Nn, 256, Mpad);

    Kin = 256;
  }

  // heads
  const float* Wt = (const float*)d_in[40];
  const float* bt = (const float*)d_in[41];
  const float* Wi = (const float*)d_in[42];
  const float* bi = (const float*)d_in[43];
  const float* Wm = (const float*)d_in[44];
  const float* bm = (const float*)d_in[45];
  float* out = (float*)d_out;
  heads_lin<<<(unsigned)((Nn * 14 + 255) / 256), 256, 0, stream>>>(
      hlay[1], Wt, bt, Wi, bi, out, Nn);
  heads_ml<<<(unsigned)((Nn * 10 + 255) / 256), 256, 0, stream>>>(
      hlay[1], out, Wm, bm, out + (size_t)Nn * 14, Nn);
}
